// RBF_euclidean_82617990906064
// MI455X (gfx1250) — matrix operand order repaired and checked by reading, not compiled, not run on hardware
//
#include <hip/hip_runtime.h>
#include <hip/hip_bf16.h>

typedef __attribute__((ext_vector_type(16))) _Float16 v16h;
typedef __attribute__((ext_vector_type(8)))  float    v8f;

#define G_PTS    1728
#define KCHUNKS  54          // G_PTS / 32
#define C_DIM    128
#define CT_TILES 8           // C_DIM / 16
#define PACK_HALVES (KCHUNKS * CT_TILES * 32 * 16)   // 221184 f16 elements

// ---------------------------------------------------------------------------
// Pack coeffs [G=1728, C=128] f32 -> f16, pre-swizzled into a contiguous-K wave32 WMMA
// B-operand layout:  packed[kc][ct][lane][h]  (h = 0..15 halves per lane)
//   lane L supplies column n = L & 15, K values K = (L<16 ? 0 : 16) + h
// so the main loop reads 32 contiguous bytes per lane (2x b128 loads).
// ---------------------------------------------------------------------------
__global__ void rbf_pack_coeffs(const float* __restrict__ coeffs,
                                _Float16* __restrict__ packed) {
    int idx = blockIdx.x * blockDim.x + threadIdx.x;
    if (idx >= PACK_HALVES) return;
    int h    =  idx        & 15;
    int lane = (idx >> 4)  & 31;
    int ct   = (idx >> 9)  & 7;
    int kc   =  idx >> 12;
    int k = kc * 32 + ((lane >> 4) << 4) + h;   // K index within G
    int n = ct * 16 + (lane & 15);              // column within C
    packed[idx] = (_Float16)coeffs[k * C_DIM + n];
}

// ---------------------------------------------------------------------------
// Fused RBF + GEMM.  Each wave: one 16-row tile of x, all 128 output columns.
// 8 waves per workgroup share the LDS-resident grid / |g|^2 tables.
// ---------------------------------------------------------------------------
__launch_bounds__(256)
__global__ void rbf_wmma_kernel(const float* __restrict__ x,
                                const float* __restrict__ grid,
                                const _Float16* __restrict__ packB,
                                const float* __restrict__ width,
                                float* __restrict__ out,
                                int N) {
    __shared__ float s_grid[G_PTS * 3];
    __shared__ float s_g2[G_PTS];

    const int tid = threadIdx.x;
    for (int g = tid; g < G_PTS; g += blockDim.x) {
        float gx = grid[g * 3 + 0];
        float gy = grid[g * 3 + 1];
        float gz = grid[g * 3 + 2];
        s_grid[g * 3 + 0] = gx;
        s_grid[g * 3 + 1] = gy;
        s_grid[g * 3 + 2] = gz;
        s_g2[g] = gx * gx + gy * gy + gz * gz;
    }
    __syncthreads();

    const float w     = width[0];
    const float invw2 = 1.0f / (w * w);        // p = exp2(-d2 * invw2)

    const int wave    = tid >> 5;
    const int lane    = tid & 31;
    const int m       = lane & 15;             // A-operand row within tile
    const int koff    = (lane >> 4) << 4;      // 0 (lanes 0-15) or 16 (lanes 16-31)
    const int rowTile = blockIdx.x * 8 + wave;
    const int row0    = rowTile * 16;

    // Load this lane's x row (clamped for the tail tile; stores are guarded).
    int rr = row0 + m;
    if (rr >= N) rr = N - 1;
    const float x0 = x[rr * 3 + 0];
    const float x1 = x[rr * 3 + 1];
    const float x2 = x[rr * 3 + 2];
    const float xsq = x0 * x0 + x1 * x1 + x2 * x2;

    v8f acc[CT_TILES] = {};

    for (int kc = 0; kc < KCHUNKS; ++kc) {
        // ---- build A tile: p[m][K] for K = kc*32 + koff + {0..15} ----
        v16h a;
        const int gbase = kc * 32 + koff;
#pragma unroll
        for (int h = 0; h < 16; ++h) {
            const int g = gbase + h;
            const float gx = s_grid[g * 3 + 0];   // broadcast within half-wave
            const float gy = s_grid[g * 3 + 1];
            const float gz = s_grid[g * 3 + 2];
            float d2 = xsq + s_g2[g] - 2.0f * (x0 * gx + x1 * gy + x2 * gz);
            d2 = fmaxf(d2, 0.0f);
            const float p = __builtin_amdgcn_exp2f(-d2 * invw2);   // v_exp_f32
            a[h] = (_Float16)p;
        }

        // ---- 8 WMMAs against the pre-swizzled B tiles (L2-resident) ----
        const _Float16* bp = packB + ((size_t)kc * CT_TILES) * 512 + lane * 16;
#pragma unroll
        for (int ct = 0; ct < CT_TILES; ++ct) {
            const v16h b = *(const v16h*)(bp + (size_t)ct * 512);
            acc[ct] = __builtin_amdgcn_wmma_f32_16x16x32_f16(
                false, a, false, b, (short)0, acc[ct], false, false);
        }
    }

    // ---- store D: lane -> column lane&15, VGPR r -> row r + 8*(lane>>4) ----
    const int mrow = (lane >> 4) << 3;
#pragma unroll
    for (int r = 0; r < 8; ++r) {
        const int orow = row0 + mrow + r;
        if (orow < N) {
            float* op = out + (size_t)orow * C_DIM + (lane & 15);
#pragma unroll
            for (int ct = 0; ct < CT_TILES; ++ct) {
                op[ct * 16] = acc[ct][r];
            }
        }
    }
}

extern "C" void kernel_launch(void* const* d_in, const int* in_sizes, int n_in,
                              void* d_out, int out_size, void* d_ws, size_t ws_size,
                              hipStream_t stream) {
    const float* x      = (const float*)d_in[0];   // [N, 3]
    const float* grid   = (const float*)d_in[1];   // [1728, 3]
    const float* coeffs = (const float*)d_in[2];   // [1728, 128]
    const float* width  = (const float*)d_in[3];   // [1]
    float* out = (float*)d_out;                    // [N, 128]
    const int N = in_sizes[0] / 3;

    _Float16* packB = (_Float16*)d_ws;             // 442368 bytes

    const int packThreads = 256;
    rbf_pack_coeffs<<<(PACK_HALVES + packThreads - 1) / packThreads,
                      packThreads, 0, stream>>>(coeffs, packB);

    const int rowTiles = (N + 15) / 16;            // 16 rows per wave
    const int wgs      = (rowTiles + 7) / 8;       // 8 waves per workgroup
    rbf_wmma_kernel<<<wgs, 256, 0, stream>>>(x, grid, packB, width, out, N);
}
